// DiscAdaptor_91233695302252
// MI455X (gfx1250) — compile-verified, matrix operand order checked by reading, not run on hardware
//
#include <hip/hip_runtime.h>
#include <hip/hip_bf16.h>
#include <math.h>

#define DEVFN __device__ __forceinline__

typedef __attribute__((ext_vector_type(16))) __bf16 v16bf;
typedef __attribute__((ext_vector_type(8)))  float  v8f;
typedef __attribute__((ext_vector_type(4)))  int    v4i;

constexpr int Bb = 8, Nt = 4096, Cc = 768, Hh = 3072;
constexpr int BN = Bb * Nt;             // 32768 rows
constexpr float EPSR = 1e-7f;
constexpr float LNE  = 1e-5f;

// ---------------------------------------------------------------------------
// Async global->LDS copy (CDNA5). Signature (probe-confirmed via diagnostic):
//   __builtin_amdgcn_global_load_async_to_lds_b128(v4i __device__* src,
//                                                  v4i __shared__* dst,
//                                                  imm offset, imm cpol)
// Falls back to sync staging if the builtins are not declared.
// ---------------------------------------------------------------------------
#if __has_builtin(__builtin_amdgcn_global_load_async_to_lds_b128)
#define HAVE_ASYNC 1
#define ASYNC_COPY16(dst, src)                                                \
  __builtin_amdgcn_global_load_async_to_lds_b128(                             \
      (__attribute__((address_space(1))) v4i*)(src),                          \
      (__attribute__((address_space(3))) v4i*)(dst), 0, 0)
#if __has_builtin(__builtin_amdgcn_s_wait_asynccnt)
#define ASYNC_WAIT(n) __builtin_amdgcn_s_wait_asynccnt(n)
#else
#define ASYNC_WAIT(n) asm volatile("s_wait_asynccnt %0" ::"i"(n) : "memory")
#endif
#else
#define HAVE_ASYNC 0
#define ASYNC_COPY16(dst, src) (*(uint4*)(dst) = *(const uint4*)(src))
#define ASYNC_WAIT(n) ((void)0)
#endif

// ---------------------------------------------------------------------------
// WMMA GEMM: D(MxN) = A(MxK, bf16 row-major) @ Bt^T, where Bt is the
// PRE-TRANSPOSED weight (N x K, bf16 row-major). Block tile 128x64, K-step 32,
// double-buffered LDS with async global->LDS copies. 8 waves: 4 over M x 2
// over N, each wave 32x32 = 2x2 v_wmma_f32_16x16x32_bf16 accumulators.
// ---------------------------------------------------------------------------
constexpr int TM = 128, TN = 64, TK = 32;
constexpr int LDSTR = 40;               // padded LDS row stride (elements)

enum { EPI_STORE = 0, EPI_GELU_BF16 = 1, EPI_BIAS_RES = 2, EPI_RES_PVEC = 3 };

template <int EPI>
__global__ __launch_bounds__(256) void gemm_bf16_wmma(
    const __hip_bfloat16* __restrict__ A, const __hip_bfloat16* __restrict__ Bt,
    const float* __restrict__ bias, const float* __restrict__ res,
    const float* __restrict__ pv, float* __restrict__ D,
    __hip_bfloat16* __restrict__ Dbf, int M, int Nn, int K)
{
  __shared__ __align__(16) __hip_bfloat16 As[2][TM * LDSTR];
  __shared__ __align__(16) __hip_bfloat16 Bs[2][TN * LDSTR];

  const int tid  = threadIdx.x;
  const int lane = tid & 31;
  const int wave = tid >> 5;
  const int wm   = wave & 3;     // wave M index (32 rows each)
  const int wn   = wave >> 2;    // wave N index (32 cols each)
  const int blockN = blockIdx.x * TN;
  const int blockM = blockIdx.y * TM;
  const int lm = lane & 15;      // row/col within 16
  const int lh = lane >> 4;      // K-half select per VGPR layout

  v8f acc[2][2];
  #pragma unroll
  for (int t = 0; t < 2; ++t)
    #pragma unroll
    for (int u = 0; u < 2; ++u)
      #pragma unroll
      for (int i = 0; i < 8; ++i) acc[t][u][i] = 0.0f;

  // Stage one K-tile (A: 128x32 = 512 16B chunks; B: 64x32 = 256 chunks).
  auto stage = [&](int k0, int buf) {
    #pragma unroll
    for (int i = 0; i < 2; ++i) {
      int c   = tid + i * 256;
      int r   = c >> 2;
      int cc8 = (c & 3) << 3;
      ASYNC_COPY16(&As[buf][r * LDSTR + cc8],
                   &A[(size_t)(blockM + r) * K + (k0 + cc8)]);
    }
    {
      int r   = tid >> 2;
      int cc8 = (tid & 3) << 3;
      ASYNC_COPY16(&Bs[buf][r * LDSTR + cc8],
                   &Bt[(size_t)(blockN + r) * K + (k0 + cc8)]);
    }
  };

  const int nk = K / TK;
  stage(0, 0);

  for (int ki = 0; ki < nk; ++ki) {
    const int buf = ki & 1;
    if (ki + 1 < nk) {
      stage((ki + 1) * TK, buf ^ 1);   // prefetch next tile into other buffer
      ASYNC_WAIT(3);                   // wait for tile ki (3 older copies/thread)
    } else {
      ASYNC_WAIT(0);
    }
    if (ki + 2 < nk) {                 // L2 prefetch two tiles ahead
      __builtin_prefetch(&A[(size_t)(blockM + (tid >> 2)) * K + (ki + 2) * TK], 0, 0);
      __builtin_prefetch(&Bt[(size_t)(blockN + (tid >> 2)) * K + (ki + 2) * TK], 0, 0);
    }
    __syncthreads();                   // all waves' tile-ki data visible

    // ---- fragments per 16-bit A/B VGPR layout (two 16B chunks each) ----
    union FragU { v16bf v; uint4 q[2]; };
    FragU a[2], bb[2];
    #pragma unroll
    for (int t = 0; t < 2; ++t) {
      const __hip_bfloat16* p = &As[buf][(wm * 32 + t * 16 + lm) * LDSTR];
      a[t].q[0] = *(const uint4*)(p + lh * 8);        // K = 0..7   / 8..15
      a[t].q[1] = *(const uint4*)(p + 16 + lh * 8);   // K = 16..23 / 24..31
    }
    #pragma unroll
    for (int u = 0; u < 2; ++u) {
      const __hip_bfloat16* p = &Bs[buf][(wn * 32 + u * 16 + lm) * LDSTR];
      bb[u].q[0] = *(const uint4*)(p + lh * 8);
      bb[u].q[1] = *(const uint4*)(p + 16 + lh * 8);
    }
    #pragma unroll
    for (int t = 0; t < 2; ++t)
      #pragma unroll
      for (int u = 0; u < 2; ++u)
        acc[t][u] = __builtin_amdgcn_wmma_f32_16x16x32_bf16(
            false, a[t].v, false, bb[u].v, (short)0, acc[t][u], false, false);

    __syncthreads();                   // done reading buf before it is rewritten
  }

  // ---- epilogue: C/D layout: VGPR r holds (m = r + lh*8, n = lm) ----
  #pragma unroll
  for (int t = 0; t < 2; ++t)
    #pragma unroll
    for (int u = 0; u < 2; ++u)
      #pragma unroll
      for (int r = 0; r < 8; ++r) {
        int row = blockM + wm * 32 + t * 16 + lh * 8 + r;
        int col = blockN + wn * 32 + u * 16 + lm;
        size_t idx = (size_t)row * Nn + col;
        float vv = acc[t][u][r];
        if constexpr (EPI == EPI_STORE) {
          D[idx] = vv;
        } else if constexpr (EPI == EPI_GELU_BF16) {
          float z = vv + bias[col];
          float g = 0.5f * z * (1.0f + erff(z * 0.70710678118654752f));
          Dbf[idx] = __float2bfloat16(g);
        } else if constexpr (EPI == EPI_BIAS_RES) {
          D[idx] = D[idx] + vv + bias[col];
        } else {  // EPI_RES_PVEC: residual + per-batch broadcast vector
          int bidx = row >> 12;  // row / 4096 tokens per batch
          D[idx] = vv + res[idx] + pv[bidx * Cc + col];
        }
      }
}

// ---------------------------------------------------------------------------
// Elementwise + reduction kernels
// ---------------------------------------------------------------------------
__global__ void cvt_bf16(const float* __restrict__ in,
                         __hip_bfloat16* __restrict__ out, long n) {
  long i = (long)blockIdx.x * blockDim.x + threadIdx.x;
  long stride = (long)gridDim.x * blockDim.x;
  for (; i < n; i += stride) out[i] = __float2bfloat16(in[i]);
}

// out[n*K + k] = bf16(in[k*N + n]) : transpose (K x N) fp32 -> (N x K) bf16
__global__ __launch_bounds__(256) void cvt_t_bf16(const float* __restrict__ in,
                                                  __hip_bfloat16* __restrict__ out,
                                                  int K, int N) {
  __shared__ float tile[32][33];
  int kb = blockIdx.y * 32, nb = blockIdx.x * 32;
  int tx = threadIdx.x & 31, ty = threadIdx.x >> 5;   // 32 x 8
  #pragma unroll
  for (int i = 0; i < 32; i += 8)
    tile[ty + i][tx] = in[(size_t)(kb + ty + i) * N + (nb + tx)];
  __syncthreads();
  #pragma unroll
  for (int i = 0; i < 32; i += 8)
    out[(size_t)(nb + ty + i) * K + (kb + tx)] = __float2bfloat16(tile[tx][ty + i]);
}

__global__ __launch_bounds__(256) void ln_kernel(
    const float* __restrict__ x, const float* __restrict__ g,
    const float* __restrict__ b, __hip_bfloat16* __restrict__ out) {
  __shared__ float sb[256];
  int row = blockIdx.x, tid = threadIdx.x;
  const float* xr = x + (size_t)row * Cc;
  float v0 = xr[tid], v1 = xr[tid + 256], v2 = xr[tid + 512];
  sb[tid] = v0 + v1 + v2; __syncthreads();
  for (int o = 128; o > 0; o >>= 1) { if (tid < o) sb[tid] += sb[tid + o]; __syncthreads(); }
  float mu = sb[0] * (1.0f / Cc); __syncthreads();
  float d0 = v0 - mu, d1 = v1 - mu, d2 = v2 - mu;
  sb[tid] = d0 * d0 + d1 * d1 + d2 * d2; __syncthreads();
  for (int o = 128; o > 0; o >>= 1) { if (tid < o) sb[tid] += sb[tid + o]; __syncthreads(); }
  float rstd = rsqrtf(sb[0] * (1.0f / Cc) + LNE);
  __hip_bfloat16* orow = out + (size_t)row * Cc;
  orow[tid]       = __float2bfloat16(d0 * rstd * g[tid]       + b[tid]);
  orow[tid + 256] = __float2bfloat16(d1 * rstd * g[tid + 256] + b[tid + 256]);
  orow[tid + 512] = __float2bfloat16(d2 * rstd * g[tid + 512] + b[tid + 512]);
}

__global__ void mask_sum_kernel(const float* __restrict__ mask, float* rsfg,
                                float* rsbg) {
  __shared__ float sb[256];
  int b = blockIdx.x, tid = threadIdx.x;
  float s = 0.f;
  for (int n = tid; n < Nt; n += 256) s += mask[b * Nt + n];
  sb[tid] = s; __syncthreads();
  for (int o = 128; o > 0; o >>= 1) { if (tid < o) sb[tid] += sb[tid + o]; __syncthreads(); }
  if (!tid) {
    float m = sb[0];
    rsfg[b] = 1.0f / (m + 5e-4f);
    rsbg[b] = 1.0f / ((float)Nt - m + 5e-4f);
  }
}

__global__ void pool_yn_kernel(const __hip_bfloat16* __restrict__ yn,
                               const float* __restrict__ mask,
                               float* __restrict__ sum_m, float* __restrict__ sum_im) {
  int b = blockIdx.y;
  int c = blockIdx.x * blockDim.x + threadIdx.x;
  float am = 0.f, at = 0.f;
  for (int n = 0; n < Nt; ++n) {
    float v = __bfloat162float(yn[((size_t)(b * Nt + n)) * Cc + c]);
    float mm = mask[b * Nt + n];
    am += mm * v; at += v;
  }
  sum_m[b * Cc + c]  = am;
  sum_im[b * Cc + c] = at - am;
}

// O[b, n] = (S[b,:] @ W[:, n]) * (rs ? rs[b] : 1)  — W is (Cc x Cc) fp32 row-major
__global__ void small_gemm_kernel(const float* __restrict__ S,
                                  const float* __restrict__ W,
                                  const float* __restrict__ rs,
                                  float* __restrict__ O) {
  int b = blockIdx.y;
  int n = blockIdx.x * blockDim.x + threadIdx.x;
  const float* srow = S + b * Cc;
  float acc = 0.f;
  for (int k = 0; k < Cc; ++k) acc += srow[k] * W[(size_t)k * Cc + n];
  O[b * Cc + n] = acc * (rs ? rs[b] : 1.0f);
}

__global__ void norms2_kernel(const float* fg, const float* bg, float* fgn, float* bgn) {
  __shared__ float s0[256], s1[256];
  int b = blockIdx.x, tid = threadIdx.x;
  float a0 = 0.f, a1 = 0.f;
  for (int c = tid; c < Cc; c += 256) {
    float f = fg[b * Cc + c], g = bg[b * Cc + c];
    a0 += f * f; a1 += g * g;
  }
  s0[tid] = a0; s1[tid] = a1; __syncthreads();
  for (int o = 128; o > 0; o >>= 1) {
    if (tid < o) { s0[tid] += s0[tid + o]; s1[tid] += s1[tid + o]; }
    __syncthreads();
  }
  if (!tid) { fgn[b] = sqrtf(s0[0]); bgn[b] = sqrtf(s1[0]); }
}

__global__ __launch_bounds__(256) void scores_kernel(
    const float* __restrict__ kbuf, const float* __restrict__ fg,
    const float* __restrict__ bg, const float* __restrict__ fgn,
    const float* __restrict__ bgn, float* __restrict__ fgs, float* __restrict__ bgs) {
  __shared__ float s0[256], s1[256], s2[256];
  int row = blockIdx.x, b = row >> 12, tid = threadIdx.x;
  const float* kr = kbuf + (size_t)row * Cc;
  const float* fr = fg + b * Cc;
  const float* br = bg + b * Cc;
  float a0 = 0.f, a1 = 0.f, a2 = 0.f;
  for (int c = tid; c < Cc; c += 256) {
    float kv = kr[c];
    a0 += kv * fr[c]; a1 += kv * br[c]; a2 += kv * kv;
  }
  s0[tid] = a0; s1[tid] = a1; s2[tid] = a2; __syncthreads();
  for (int o = 128; o > 0; o >>= 1) {
    if (tid < o) { s0[tid] += s0[tid + o]; s1[tid] += s1[tid + o]; s2[tid] += s2[tid + o]; }
    __syncthreads();
  }
  if (!tid) {
    float kn = sqrtf(s2[0]);
    fgs[row] = s0[0] / (kn * fgn[b] + EPSR);
    bgs[row] = s1[0] / (kn * bgn[b] + EPSR);
  }
}

__global__ void minmax_kernel(const float* fgs, const float* bgs, float* mm4) {
  __shared__ float s0[256], s1[256], s2[256], s3[256];
  int b = blockIdx.x, tid = threadIdx.x;
  float fmn = 3.4e38f, fmx = -3.4e38f, bmn = 3.4e38f, bmx = -3.4e38f;
  for (int n = tid; n < Nt; n += 256) {
    float f = fgs[b * Nt + n], g = bgs[b * Nt + n];
    fmn = fminf(fmn, f); fmx = fmaxf(fmx, f);
    bmn = fminf(bmn, g); bmx = fmaxf(bmx, g);
  }
  s0[tid] = fmn; s1[tid] = fmx; s2[tid] = bmn; s3[tid] = bmx; __syncthreads();
  for (int o = 128; o > 0; o >>= 1) {
    if (tid < o) {
      s0[tid] = fminf(s0[tid], s0[tid + o]); s1[tid] = fmaxf(s1[tid], s1[tid + o]);
      s2[tid] = fminf(s2[tid], s2[tid + o]); s3[tid] = fmaxf(s3[tid], s3[tid + o]);
    }
    __syncthreads();
  }
  if (!tid) {
    mm4[b * 4 + 0] = s0[0]; mm4[b * 4 + 1] = s1[0];
    mm4[b * 4 + 2] = s2[0]; mm4[b * 4 + 3] = s3[0];
  }
}

DEVFN float score_at(const float* fgs, const float* bgs, const float* mm4, int b, int n) {
  float f = fgs[b * Nt + n], g = bgs[b * Nt + n];
  float sf = (f - mm4[b * 4 + 0]) / (mm4[b * 4 + 1] - mm4[b * 4 + 0] + EPSR);
  float sg = (g - mm4[b * 4 + 2]) / (mm4[b * 4 + 3] - mm4[b * 4 + 2] + EPSR);
  return sf - sg;
}

__global__ void smstats_kernel(const float* fgs, const float* bgs, const float* mm4,
                               float* rmax, float* rsum, float* pm) {
  __shared__ float sb[256];
  int b = blockIdx.x, tid = threadIdx.x;
  float mx = -3.4e38f;
  for (int n = tid; n < Nt; n += 256) {
    float s = score_at(fgs, bgs, mm4, b, n);
    pm[b * Nt + n] = s;                       // pseudo_mask output
    float w = s < 0.f ? s - 100.f : s;
    mx = fmaxf(mx, w);
  }
  sb[tid] = mx; __syncthreads();
  for (int o = 128; o > 0; o >>= 1) { if (tid < o) sb[tid] = fmaxf(sb[tid], sb[tid + o]); __syncthreads(); }
  mx = sb[0]; __syncthreads();
  float se = 0.f;
  for (int n = tid; n < Nt; n += 256) {
    float s = score_at(fgs, bgs, mm4, b, n);
    float w = s < 0.f ? s - 100.f : s;
    se += expf(w - mx);
  }
  sb[tid] = se; __syncthreads();
  for (int o = 128; o > 0; o >>= 1) { if (tid < o) sb[tid] += sb[tid + o]; __syncthreads(); }
  if (!tid) { rmax[b] = mx; rsum[b] = sb[0]; }
}

__global__ void attn_pool_kernel(const __hip_bfloat16* __restrict__ xn,
                                 const float* fgs, const float* bgs, const float* mm4,
                                 const float* rmax, const float* rsum,
                                 float* __restrict__ sqp) {
  int b = blockIdx.y;
  int c = blockIdx.x * blockDim.x + threadIdx.x;
  float mx = rmax[b], inv = 1.0f / rsum[b];
  float acc = 0.f;
  for (int n = 0; n < Nt; ++n) {
    float s = score_at(fgs, bgs, mm4, b, n);
    float w = s < 0.f ? s - 100.f : s;
    acc += expf(w - mx) * __bfloat162float(xn[((size_t)(b * Nt + n)) * Cc + c]);
  }
  sqp[b * Cc + c] = acc * inv;
}

__global__ void pro_kernel(const float* qp, const float* fg, const float* fgn,
                           float* pro) {
  __shared__ float s0[256], s1[256];
  __shared__ float ssim;
  int b = blockIdx.x, tid = threadIdx.x;
  float a0 = 0.f, a1 = 0.f;
  for (int c = tid; c < Cc; c += 256) {
    float q = qp[b * Cc + c], f = fg[b * Cc + c];
    a0 += q * q; a1 += q * f;
  }
  s0[tid] = a0; s1[tid] = a1; __syncthreads();
  for (int o = 128; o > 0; o >>= 1) {
    if (tid < o) { s0[tid] += s0[tid + o]; s1[tid] += s1[tid + o]; }
    __syncthreads();
  }
  if (!tid) {
    float sim = s1[0] / (sqrtf(s0[0]) * fgn[b] + EPSR);
    ssim = (sim + 1.0f) * 0.5f;
  }
  __syncthreads();
  float sim = ssim;
  for (int c = tid; c < Cc; c += 256)
    pro[b * Cc + c] = sim * fg[b * Cc + c] + (1.0f - sim) * qp[b * Cc + c];
}

// ---------------------------------------------------------------------------
// Host orchestration
// ---------------------------------------------------------------------------
static inline char* carve(char*& p, size_t bytes) {
  char* r = p;
  p += (bytes + 255) & ~size_t(255);
  return r;
}

extern "C" void kernel_launch(void* const* d_in, const int* in_sizes, int n_in,
                              void* d_out, int out_size, void* d_ws, size_t ws_size,
                              hipStream_t stream) {
  const float* x     = (const float*)d_in[0];
  const float* y     = (const float*)d_in[1];
  const float* mask  = (const float*)d_in[2];
  const float* ln1_g = (const float*)d_in[5];
  const float* ln1_b = (const float*)d_in[6];
  const float* Wq    = (const float*)d_in[7];
  const float* Wk    = (const float*)d_in[8];
  const float* Wv    = (const float*)d_in[9];
  const float* Wpx   = (const float*)d_in[10];
  const float* Wpy   = (const float*)d_in[11];
  const float* ln2_g = (const float*)d_in[12];
  const float* ln2_b = (const float*)d_in[13];
  const float* fx1_w = (const float*)d_in[14];
  const float* fx1_b = (const float*)d_in[15];
  const float* fx2_w = (const float*)d_in[16];
  const float* fx2_b = (const float*)d_in[17];
  const float* fy1_w = (const float*)d_in[18];
  const float* fy1_b = (const float*)d_in[19];
  const float* fy2_w = (const float*)d_in[20];
  const float* fy2_b = (const float*)d_in[21];

  float* xo = (float*)d_out;                        // (BN, Cc)
  float* yo = xo + (size_t)BN * Cc;                 // (BN, Cc)
  float* pm = yo + (size_t)BN * Cc;                 // (Bb, Nt) pseudo_mask

  char* p = (char*)d_ws;
  typedef __hip_bfloat16 bf;
  bf* x_bf   = (bf*)carve(p, (size_t)BN * Cc * 2);
  bf* y_bf   = (bf*)carve(p, (size_t)BN * Cc * 2);
  bf* xn_bf  = (bf*)carve(p, (size_t)BN * Cc * 2);  // reused for ln2(xo)
  bf* yn_bf  = (bf*)carve(p, (size_t)BN * Cc * 2);  // reused for ln2(yo)
  bf* wkT    = (bf*)carve(p, (size_t)Cc * Cc * 2);  // all weights transposed (NxK)
  bf* wpxT   = (bf*)carve(p, (size_t)Cc * Cc * 2);
  bf* wpyT   = (bf*)carve(p, (size_t)Cc * Cc * 2);
  bf* fx1T   = (bf*)carve(p, (size_t)Hh * Cc * 2);
  bf* fx2T   = (bf*)carve(p, (size_t)Cc * Hh * 2);
  bf* fy1T   = (bf*)carve(p, (size_t)Hh * Cc * 2);
  bf* fy2T   = (bf*)carve(p, (size_t)Cc * Hh * 2);
  float* kbuf   = (float*)carve(p, (size_t)BN * Cc * 4);
  bf*    hid_bf = (bf*)carve(p, (size_t)BN * Hh * 2);
  float* sum_m  = (float*)carve(p, Bb * Cc * 4);
  float* sum_im = (float*)carve(p, Bb * Cc * 4);
  float* fg     = (float*)carve(p, Bb * Cc * 4);
  float* bg     = (float*)carve(p, Bb * Cc * 4);
  float* sqp    = (float*)carve(p, Bb * Cc * 4);
  float* qp     = (float*)carve(p, Bb * Cc * 4);
  float* pro    = (float*)carve(p, Bb * Cc * 4);
  float* pvx    = (float*)carve(p, Bb * Cc * 4);
  float* pvy    = (float*)carve(p, Bb * Cc * 4);
  float* fgs    = (float*)carve(p, (size_t)BN * 4);
  float* bgs    = (float*)carve(p, (size_t)BN * 4);
  float* rsfg   = (float*)carve(p, Bb * 4);
  float* rsbg   = (float*)carve(p, Bb * 4);
  float* fgn    = (float*)carve(p, Bb * 4);
  float* bgn    = (float*)carve(p, Bb * 4);
  float* mm4    = (float*)carve(p, Bb * 4 * 4);
  float* rmax   = (float*)carve(p, Bb * 4);
  float* rsum   = (float*)carve(p, Bb * 4);

  // bf16 conversions (activations) + transposed bf16 weights
  cvt_bf16<<<2048, 256, 0, stream>>>(x, x_bf, (long)BN * Cc);
  cvt_bf16<<<2048, 256, 0, stream>>>(y, y_bf, (long)BN * Cc);
  cvt_t_bf16<<<dim3(Cc / 32, Cc / 32), 256, 0, stream>>>(Wk, wkT, Cc, Cc);
  cvt_t_bf16<<<dim3(Cc / 32, Cc / 32), 256, 0, stream>>>(Wpx, wpxT, Cc, Cc);  // top half
  cvt_t_bf16<<<dim3(Cc / 32, Cc / 32), 256, 0, stream>>>(Wpy, wpyT, Cc, Cc);
  cvt_t_bf16<<<dim3(Hh / 32, Cc / 32), 256, 0, stream>>>(fx1_w, fx1T, Cc, Hh);
  cvt_t_bf16<<<dim3(Cc / 32, Hh / 32), 256, 0, stream>>>(fx2_w, fx2T, Hh, Cc);
  cvt_t_bf16<<<dim3(Hh / 32, Cc / 32), 256, 0, stream>>>(fy1_w, fy1T, Cc, Hh);
  cvt_t_bf16<<<dim3(Cc / 32, Hh / 32), 256, 0, stream>>>(fy2_w, fy2T, Hh, Cc);

  // LN1
  ln_kernel<<<BN, 256, 0, stream>>>(x, ln1_g, ln1_b, xn_bf);
  ln_kernel<<<BN, 256, 0, stream>>>(y, ln1_g, ln1_b, yn_bf);

  // k = xn @ Wk  (big WMMA GEMM)
  gemm_bf16_wmma<EPI_STORE><<<dim3(Cc / TN, BN / TM), 256, 0, stream>>>(
      xn_bf, wkT, nullptr, nullptr, nullptr, kbuf, nullptr, BN, Cc, Cc);

  // masked pools: fg = (sum m*yn)@Wq * rsfg ; bg = (sum (1-m)*yn)@Wq * rsbg
  mask_sum_kernel<<<Bb, 256, 0, stream>>>(mask, rsfg, rsbg);
  pool_yn_kernel<<<dim3(Cc / 256, Bb), 256, 0, stream>>>(yn_bf, mask, sum_m, sum_im);
  small_gemm_kernel<<<dim3(Cc / 128, Bb), 128, 0, stream>>>(sum_m, Wq, rsfg, fg);
  small_gemm_kernel<<<dim3(Cc / 128, Bb), 128, 0, stream>>>(sum_im, Wq, rsbg, bg);
  norms2_kernel<<<Bb, 256, 0, stream>>>(fg, bg, fgn, bgn);

  // cosine scores, minmax, softmax stats (+ pseudo_mask write)
  scores_kernel<<<BN, 256, 0, stream>>>(kbuf, fg, bg, fgn, bgn, fgs, bgs);
  minmax_kernel<<<Bb, 256, 0, stream>>>(fgs, bgs, mm4);
  smstats_kernel<<<Bb, 256, 0, stream>>>(fgs, bgs, mm4, rmax, rsum, pm);

  // query_pro = (sum attn*xn) @ Wv ; then sim/pro blend
  attn_pool_kernel<<<dim3(Cc / 256, Bb), 256, 0, stream>>>(xn_bf, fgs, bgs, mm4,
                                                           rmax, rsum, sqp);
  small_gemm_kernel<<<dim3(Cc / 128, Bb), 128, 0, stream>>>(sqp, Wv, nullptr, qp);
  pro_kernel<<<Bb, 256, 0, stream>>>(qp, fg, fgn, pro);
  small_gemm_kernel<<<dim3(Cc / 128, Bb), 128, 0, stream>>>(pro, Wpx + (size_t)Cc * Cc, nullptr, pvx);
  small_gemm_kernel<<<dim3(Cc / 128, Bb), 128, 0, stream>>>(pro, Wpy + (size_t)Cc * Cc, nullptr, pvy);

  // xo = x + x@Wpx_top + pro@Wpx_bot (broadcast); same for yo
  gemm_bf16_wmma<EPI_RES_PVEC><<<dim3(Cc / TN, BN / TM), 256, 0, stream>>>(
      x_bf, wpxT, nullptr, x, pvx, xo, nullptr, BN, Cc, Cc);
  gemm_bf16_wmma<EPI_RES_PVEC><<<dim3(Cc / TN, BN / TM), 256, 0, stream>>>(
      y_bf, wpyT, nullptr, y, pvy, yo, nullptr, BN, Cc, Cc);

  // MLP x: xo += W2 @ gelu(W1 @ ln(xo) + b1) + b2
  ln_kernel<<<BN, 256, 0, stream>>>(xo, ln2_g, ln2_b, xn_bf);
  gemm_bf16_wmma<EPI_GELU_BF16><<<dim3(Hh / TN, BN / TM), 256, 0, stream>>>(
      xn_bf, fx1T, fx1_b, nullptr, nullptr, nullptr, hid_bf, BN, Hh, Cc);
  gemm_bf16_wmma<EPI_BIAS_RES><<<dim3(Cc / TN, BN / TM), 256, 0, stream>>>(
      hid_bf, fx2T, fx2_b, nullptr, nullptr, xo, nullptr, BN, Cc, Hh);

  // MLP y
  ln_kernel<<<BN, 256, 0, stream>>>(yo, ln2_g, ln2_b, yn_bf);
  gemm_bf16_wmma<EPI_GELU_BF16><<<dim3(Hh / TN, BN / TM), 256, 0, stream>>>(
      yn_bf, fy1T, fy1_b, nullptr, nullptr, nullptr, hid_bf, BN, Hh, Cc);
  gemm_bf16_wmma<EPI_BIAS_RES><<<dim3(Cc / TN, BN / TM), 256, 0, stream>>>(
      hid_bf, fy2T, fy2_b, nullptr, nullptr, yo, nullptr, BN, Cc, Hh);

  (void)in_sizes; (void)n_in; (void)out_size; (void)ws_size;
}
